// ImprovedEdgeGNN_60189671686718
// MI455X (gfx1250) — compile-verified, matrix operand order checked by reading, not run on hardware
//
#include <hip/hip_runtime.h>
#include <hip/hip_bf16.h>

typedef _Float16 v16h __attribute__((ext_vector_type(16)));
typedef float    v8f  __attribute__((ext_vector_type(8)));
typedef _Float16 h8   __attribute__((ext_vector_type(8)));

constexpr int kB = 8, kN = 512, kD = 512, kH = 256, kE = 32, kC = 2;

#define GAMMA   (-0.1f)
#define ZETA    (1.1f)
#define LAMBDA  (0.01f)
// BETA*log(-GAMMA/ZETA) = 0.66 * log(0.1/1.1) = -1.5826109...
#define PEN_SHIFT (1.5826109f)

// ---------------------------------------------------------------------------
// Kernel A: per-row L2 normalize, write transposed f16 features, compute e1/e2
// grid = B*N blocks, 256 threads
// ---------------------------------------------------------------------------
__global__ void k_norm_edge(const float* __restrict__ node_feat,
                            const float* __restrict__ Wa,
                            const float* __restrict__ Wb,
                            _Float16* __restrict__ nfT,   // [B][d*512 + n]
                            float* __restrict__ e1,       // [B*N, 32]
                            float* __restrict__ e2) {
  const int idx = blockIdx.x;            // b*512 + n
  const int b = idx >> 9, n = idx & 511;
  const int t = threadIdx.x;
  __shared__ float row[kD];
  __shared__ float red[256];
  __shared__ float sInv;

  float x1 = node_feat[(size_t)idx * kD + t];
  float x2 = node_feat[(size_t)idx * kD + t + 256];
  row[t] = x1; row[t + 256] = x2;
  red[t] = x1 * x1 + x2 * x2;
  __syncthreads();
  for (int o = 128; o > 0; o >>= 1) {
    if (t < o) red[t] += red[t + o];
    __syncthreads();
  }
  if (t == 0) sInv = 1.0f / fmaxf(sqrtf(red[0]), 1e-12f);
  __syncthreads();
  const float inv = sInv;

  // transposed f16 copy: nfT[b][d*512 + n]
  nfT[(size_t)b * kD * kN + (size_t)t * kN + n]          = (_Float16)(row[t] * inv);
  nfT[(size_t)b * kD * kN + (size_t)(t + 256) * kN + n]  = (_Float16)(row[t + 256] * inv);
  __syncthreads();

  // 64 outputs (e1: 0..31, e2: 32..63), 4 threads each over d
  const int e = t >> 2;
  const int part = t & 3;
  const int ec = e & 31;
  const float* __restrict__ Wp = (e < 32) ? Wa : Wb;
  float s = 0.f;
  const int d0 = part * 128;
  #pragma unroll 4
  for (int d = d0; d < d0 + 128; ++d)
    s += row[d] * inv * Wp[d * kE + ec];
  red[t] = s;
  __syncthreads();
  if (part == 0) {
    float tot = red[t] + red[t + 1] + red[t + 2] + red[t + 3];
    if (e < 32) e1[(size_t)idx * kE + ec] = tot;
    else        e2[(size_t)idx * kE + ec] = tot;
  }
}

// ---------------------------------------------------------------------------
// Kernel B: pairwise edge MLP + hard-concrete gate -> wadj (into d_out),
// per-block L0-penalty partial sums.  grid = (32, 32, 8), 256 threads
// ---------------------------------------------------------------------------
__global__ void k_pairwise(const float* __restrict__ e1,
                           const float* __restrict__ e2,
                           const float* __restrict__ be,
                           const float* __restrict__ W2e,
                           const float* __restrict__ b2e,
                           const float* __restrict__ adjs,
                           float* __restrict__ wadj_out,   // d_out + 17
                           float* __restrict__ partials) {
  const int i0 = blockIdx.x * 16, j0 = blockIdx.y * 16, b = blockIdx.z;
  const int t = threadIdx.x;
  const int ti = t >> 4, tj = t & 15;
  __shared__ float se1[16 * kE];
  __shared__ float se2[16 * kE];
  __shared__ float sw[kE], sb[kE];
  __shared__ float red[256];

  const size_t base1 = ((size_t)b * kN + i0) * kE;
  const size_t base2 = ((size_t)b * kN + j0) * kE;
  se1[t] = e1[base1 + t]; se1[t + 256] = e1[base1 + t + 256];
  se2[t] = e2[base2 + t]; se2[t + 256] = e2[base2 + t + 256];
  if (t < kE) { sw[t] = W2e[t]; sb[t] = be[t]; }
  __syncthreads();

  float acc = 0.f;
  #pragma unroll
  for (int e = 0; e < kE; ++e) {
    float s = se1[ti * kE + e] + se2[tj * kE + e] + sb[e];
    acc += fmaxf(s, 0.f) * sw[e];
  }
  const float la = acc + b2e[0];
  const float sig = 1.0f / (1.0f + expf(-la));
  const float gate = fminf(fmaxf(sig * (ZETA - GAMMA) + GAMMA, 0.f), 1.f);
  const size_t off = (size_t)b * kN * kN + (size_t)(i0 + ti) * kN + (j0 + tj);
  const float a = adjs[off];
  wadj_out[off] = gate * a * a;

  red[t] = 1.0f / (1.0f + expf(-(la + PEN_SHIFT)));
  __syncthreads();
  for (int o = 128; o > 0; o >>= 1) {
    if (t < o) red[t] += red[t + o];
    __syncthreads();
  }
  if (t == 0) {
    int bi = (blockIdx.z * gridDim.x + blockIdx.x) * gridDim.y + blockIdx.y;
    partials[bi] = red[0];
  }
}

// ---------------------------------------------------------------------------
// Kernel C: row sums of wadj. grid = B*N blocks, 256 threads
// ---------------------------------------------------------------------------
__global__ void k_rowsum(const float* __restrict__ wadj, float* __restrict__ sums) {
  const int r = blockIdx.x;
  const int t = threadIdx.x;
  __shared__ float red[256];
  red[t] = wadj[(size_t)r * kN + t] + wadj[(size_t)r * kN + t + 256];
  __syncthreads();
  for (int o = 128; o > 0; o >>= 1) {
    if (t < o) red[t] += red[t + o];
    __syncthreads();
  }
  if (t == 0) sums[r] = red[0];
}

// ---------------------------------------------------------------------------
// Kernel D: A16 = f16(wadj / (rowsum + 1e-8)). grid = 8192, 256 threads
// ---------------------------------------------------------------------------
__global__ void k_scale_f16(const float* __restrict__ wadj,
                            const float* __restrict__ sums,
                            _Float16* __restrict__ A16) {
  const size_t idx = (size_t)blockIdx.x * 256 + threadIdx.x;
  const int r = (int)(idx >> 9);
  A16[idx] = (_Float16)(wadj[idx] * (1.0f / (sums[r] + 1e-8f)));
}

// ---------------------------------------------------------------------------
// Kernel P: WconvT16[h*512 + d] = f16(Wconv[d*256 + h]). grid = 512, 256 thr
// ---------------------------------------------------------------------------
__global__ void k_wconvT(const float* __restrict__ Wconv, _Float16* __restrict__ WT) {
  const size_t idx = (size_t)blockIdx.x * 256 + threadIdx.x;
  const int d = (int)(idx >> 8);       // /256
  const int h = (int)(idx & 255);
  WT[(size_t)h * kD + d] = (_Float16)Wconv[idx];
}

// ---------------------------------------------------------------------------
// WMMA K-loop: C(16x16,f32) += A(16x32,f16) x B(32x16,f16), K in steps of 32.
// Arow = row pointer for this lane's M row (A row-major, stride lda implied by
// caller), Brow = row pointer for this lane's N column in the transposed B
// buffer (Bt row-major N x K). Fragment element order follows the CDNA5
// 16-bit A 16x32 layout: lanes 0-15 hold K {0..7,16..23}, lanes 16-31 hold
// K {8..15,24..31} of each 32-wide K block; B mirrors it via the Bt buffer.
// ---------------------------------------------------------------------------
__device__ __forceinline__ v8f wmma_kloop(const _Float16* __restrict__ Arow,
                                          const _Float16* __restrict__ Brow,
                                          int K, int half) {
  v8f acc = {0.f, 0.f, 0.f, 0.f, 0.f, 0.f, 0.f, 0.f};
  for (int k = 0; k < K; k += 32) {
    const int kb = k + half * 8;
    __builtin_prefetch(Arow + kb + 32, 0, 3);
    __builtin_prefetch(Brow + kb + 32, 0, 3);
    h8 a0 = *(const h8*)(Arow + kb);
    h8 a1 = *(const h8*)(Arow + kb + 16);
    h8 b0 = *(const h8*)(Brow + kb);
    h8 b1 = *(const h8*)(Brow + kb + 16);
    v16h av, bv;
    #pragma unroll
    for (int i = 0; i < 8; ++i) {
      av[i] = a0[i]; av[i + 8] = a1[i];
      bv[i] = b0[i]; bv[i + 8] = b1[i];
    }
    acc = __builtin_amdgcn_wmma_f32_16x16x32_f16(
        /*neg_a=*/false, av, /*neg_b=*/false, bv,
        /*c_mod=*/(short)0, acc, /*reuse_a=*/false, /*reuse_b=*/false);
  }
  return acc;
}

// ---------------------------------------------------------------------------
// Kernel E: agg16 = A16(norm wadj) @ nf   (per batch, M=N=K=512), f16 out.
// grid = (8, 16, B); 256 threads = 8 waves, wave tile grid 4x2 -> 64x32/block
// ---------------------------------------------------------------------------
__global__ void k_gemm_agg(const _Float16* __restrict__ A16,
                           const _Float16* __restrict__ nfT,
                           _Float16* __restrict__ agg16) {
  const int b = blockIdx.z;
  const int lane = threadIdx.x & 31;
  const int wave = threadIdx.x >> 5;
  const int wr = wave & 3, wc = wave >> 2;
  const int r0 = blockIdx.x * 64 + wr * 16;
  const int c0 = blockIdx.y * 32 + wc * 16;
  const int half = lane >> 4, l15 = lane & 15;

  const _Float16* A  = A16  + (size_t)b * kN * kN;
  const _Float16* Bt = nfT  + (size_t)b * kD * kN;   // Bt[d*512 + j]
  _Float16*       Cc = agg16 + (size_t)b * kN * kD;

  v8f acc = wmma_kloop(A + (size_t)(r0 + l15) * kN,
                       Bt + (size_t)(c0 + l15) * kN, kN, half);

  const int col = c0 + l15;
  const int rbase = r0 + half * 8;
  #pragma unroll
  for (int v = 0; v < 8; ++v)
    Cc[(size_t)(rbase + v) * kD + col] = (_Float16)acc[v];
}

// ---------------------------------------------------------------------------
// Kernel F: h = relu(agg @ Wconv + bconv)  (per batch, M=512 K=512 N=256)
// grid = (8, 8, B); 256 threads
// ---------------------------------------------------------------------------
__global__ void k_gemm_conv(const _Float16* __restrict__ agg16,
                            const _Float16* __restrict__ WconvT,
                            const float* __restrict__ bconv,
                            float* __restrict__ hbuf) {
  const int b = blockIdx.z;
  const int lane = threadIdx.x & 31;
  const int wave = threadIdx.x >> 5;
  const int wr = wave & 3, wc = wave >> 2;
  const int r0 = blockIdx.x * 64 + wr * 16;
  const int c0 = blockIdx.y * 32 + wc * 16;
  const int half = lane >> 4, l15 = lane & 15;

  const _Float16* A  = agg16 + (size_t)b * kN * kD;
  float*          Cc = hbuf  + (size_t)b * kN * kH;

  v8f acc = wmma_kloop(A + (size_t)(r0 + l15) * kD,
                       WconvT + (size_t)(c0 + l15) * kD, kD, half);

  const int col = c0 + l15;
  const int rbase = r0 + half * 8;
  const float bias = bconv[col];
  #pragma unroll
  for (int v = 0; v < 8; ++v)
    Cc[(size_t)(rbase + v) * kH + col] = fmaxf(acc[v] + bias, 0.f);
}

// ---------------------------------------------------------------------------
// Kernel R: fixed-order reduce of penalty partials -> one scalar
// ---------------------------------------------------------------------------
__global__ void k_reduce_pen(const float* __restrict__ partials, int n,
                             float* __restrict__ pen) {
  const int t = threadIdx.x;
  __shared__ float red[256];
  float s = 0.f;
  for (int i = t; i < n; i += 256) s += partials[i];
  red[t] = s;
  __syncthreads();
  for (int o = 128; o > 0; o >>= 1) {
    if (t < o) red[t] += red[t + o];
    __syncthreads();
  }
  if (t == 0) pen[0] = red[0];
}

// ---------------------------------------------------------------------------
// Kernel G: masked softmax over weighted degree, attention pooling.
// grid = B blocks, 256 threads (thread t == hidden channel t)
// ---------------------------------------------------------------------------
__global__ void k_pool(const float* __restrict__ sums,
                       const float* __restrict__ masks,
                       const float* __restrict__ hbuf,
                       float* __restrict__ gr) {
  const int b = blockIdx.x;
  const int t = threadIdx.x;
  __shared__ float attn[kN];
  __shared__ float red[256];
  __shared__ float scal;

  const float m1 = masks[(size_t)b * kN + t];
  const float m2 = masks[(size_t)b * kN + t + 256];
  const float l1 = (m1 > 0.f) ? sums[(size_t)b * kN + t]       : -1e9f;
  const float l2 = (m2 > 0.f) ? sums[(size_t)b * kN + t + 256] : -1e9f;
  red[t] = fmaxf(l1, l2);
  __syncthreads();
  for (int o = 128; o > 0; o >>= 1) {
    if (t < o) red[t] = fmaxf(red[t], red[t + o]);
    __syncthreads();
  }
  if (t == 0) scal = red[0];
  __syncthreads();
  const float mx = scal;
  const float ex1 = expf(l1 - mx), ex2 = expf(l2 - mx);
  attn[t] = ex1; attn[t + 256] = ex2;
  red[t] = ex1 + ex2;
  __syncthreads();
  for (int o = 128; o > 0; o >>= 1) {
    if (t < o) red[t] += red[t + o];
    __syncthreads();
  }
  if (t == 0) scal = 1.0f / red[0];
  __syncthreads();
  const float inv = scal;
  attn[t] *= inv; attn[t + 256] *= inv;
  __syncthreads();

  float acc = 0.f;
  const float* hb = hbuf + (size_t)b * kN * kH + t;
  for (int n = 0; n < kN; ++n) acc += attn[n] * hb[(size_t)n * kH];
  gr[b * kH + t] = acc;
}

// ---------------------------------------------------------------------------
// Kernel H: classifier (relu -> layernorm -> spectral-norm linear), CE loss,
// total loss. 1 block, 128 threads.
// ---------------------------------------------------------------------------
__global__ void k_classifier(const float* __restrict__ gr,
                             const float* __restrict__ W1c,
                             const float* __restrict__ b1c,
                             const float* __restrict__ ln_g,
                             const float* __restrict__ ln_b,
                             const float* __restrict__ W2c,
                             const float* __restrict__ b2c,
                             const float* __restrict__ u_sn,
                             const int* __restrict__ labels,
                             const float* __restrict__ pen,
                             float* __restrict__ d_out) {
  const int t = threadIdx.x;           // 0..127
  const int Hh = kH / 2;               // 128
  __shared__ float z[kB][kH / 2];
  __shared__ float red[128];
  __shared__ float sc[8];
  __shared__ float lg[kB * kC];

  // z = relu(gr @ W1c + b1c)
  for (int b = 0; b < kB; ++b) {
    float s = b1c[t];
    for (int hh = 0; hh < kH; ++hh) s += gr[b * kH + hh] * W1c[hh * Hh + t];
    z[b][t] = fmaxf(s, 0.f);
  }
  __syncthreads();

  // layernorm per row
  for (int b = 0; b < kB; ++b) {
    red[t] = z[b][t];
    __syncthreads();
    for (int o = 64; o > 0; o >>= 1) { if (t < o) red[t] += red[t + o]; __syncthreads(); }
    if (t == 0) sc[0] = red[0] / (float)Hh;
    __syncthreads();
    const float mu = sc[0];
    const float d = z[b][t] - mu;
    red[t] = d * d;
    __syncthreads();
    for (int o = 64; o > 0; o >>= 1) { if (t < o) red[t] += red[t + o]; __syncthreads(); }
    if (t == 0) sc[1] = red[0] / (float)Hh;
    __syncthreads();
    z[b][t] = d * rsqrtf(sc[1] + 1e-5f) * ln_g[t] + ln_b[t];
    __syncthreads();
  }

  // spectral norm: one power iteration
  const float u0 = u_sn[0], u1 = u_sn[1];
  const float w0 = W2c[t * 2], w1 = W2c[t * 2 + 1];
  float v = w0 * u0 + w1 * u1;
  red[t] = v * v;
  __syncthreads();
  for (int o = 64; o > 0; o >>= 1) { if (t < o) red[t] += red[t + o]; __syncthreads(); }
  if (t == 0) sc[0] = fmaxf(sqrtf(red[0]), 1e-12f);
  __syncthreads();
  const float vh = v / sc[0];
  red[t] = w0 * vh;
  __syncthreads();
  for (int o = 64; o > 0; o >>= 1) { if (t < o) red[t] += red[t + o]; __syncthreads(); }
  if (t == 0) sc[1] = red[0];
  __syncthreads();
  red[t] = w1 * vh;
  __syncthreads();
  for (int o = 64; o > 0; o >>= 1) { if (t < o) red[t] += red[t + o]; __syncthreads(); }
  if (t == 0) {
    sc[2] = red[0];
    float n2 = fmaxf(sqrtf(sc[1] * sc[1] + sc[2] * sc[2]), 1e-12f);
    sc[3] = sc[1] / n2; sc[4] = sc[2] / n2;
  }
  __syncthreads();
  red[t] = vh * (w0 * sc[3] + w1 * sc[4]);
  __syncthreads();
  for (int o = 64; o > 0; o >>= 1) { if (t < o) red[t] += red[t + o]; __syncthreads(); }
  if (t == 0) sc[5] = red[0];                        // sigma
  __syncthreads();
  const float sigma = sc[5];

  // logits
  if (t < kB * kC) {
    const int b = t >> 1, c = t & 1;
    float s = 0.f;
    for (int k = 0; k < Hh; ++k) s += z[b][k] * W2c[k * 2 + c];
    s = s / sigma + b2c[c];
    lg[t] = s;
    d_out[t] = s;
  }
  __syncthreads();

  // losses
  if (t == 0) {
    float cls = 0.f;
    for (int b = 0; b < kB; ++b) {
      const float l0 = lg[b * 2], l1 = lg[b * 2 + 1];
      const float m = fmaxf(l0, l1);
      const float lse = m + logf(expf(l0 - m) + expf(l1 - m));
      const int lab = labels[b];
      cls += -(((lab == 0) ? l0 : l1) - lse);
    }
    cls /= (float)kB;
    d_out[kB * kC] = cls + LAMBDA * pen[0];
  }
}

// ---------------------------------------------------------------------------
// launch
// ---------------------------------------------------------------------------
extern "C" void kernel_launch(void* const* d_in, const int* in_sizes, int n_in,
                              void* d_out, int out_size, void* d_ws, size_t ws_size,
                              hipStream_t stream) {
  const float* node_feat = (const float*)d_in[0];
  const int*   labels    = (const int*)  d_in[1];
  const float* adjs      = (const float*)d_in[2];
  const float* masks     = (const float*)d_in[3];
  const float* Wa        = (const float*)d_in[4];
  const float* Wb        = (const float*)d_in[5];
  const float* be        = (const float*)d_in[6];
  const float* W2e       = (const float*)d_in[7];
  const float* b2e       = (const float*)d_in[8];
  const float* Wconv     = (const float*)d_in[9];
  const float* bconv     = (const float*)d_in[10];
  const float* W1c       = (const float*)d_in[11];
  const float* b1c       = (const float*)d_in[12];
  const float* ln_g      = (const float*)d_in[13];
  const float* ln_b      = (const float*)d_in[14];
  const float* W2c       = (const float*)d_in[15];
  const float* b2c       = (const float*)d_in[16];
  const float* u_sn      = (const float*)d_in[17];

  float* out = (float*)d_out;
  float* wadj = out + kB * kC + 1;     // third output: adjs*edge_weights

  // workspace layout (all offsets 256B-aligned)
  char* ws = (char*)d_ws;
  float*    e1       = (float*)   (ws + 0);                 //  512 KB
  float*    e2       = (float*)   (ws + 524288);            //  512 KB
  _Float16* nfT      = (_Float16*)(ws + 1048576);           //    4 MB
  float*    sums     = (float*)   (ws + 5242880);           //   16 KB
  _Float16* A16      = (_Float16*)(ws + 5259264);           //    4 MB
  _Float16* agg16    = (_Float16*)(ws + 9453568);           //    4 MB
  float*    hbuf     = (float*)   (ws + 13647872);          //    4 MB
  float*    gr       = (float*)   (ws + 17842176);          //    8 KB
  float*    partials = (float*)   (ws + 17850368);          //   32 KB
  float*    pen      = (float*)   (ws + 17883136);          //  scalar
  _Float16* WconvT   = (_Float16*)(ws + 17883392);          //  256 KB

  k_norm_edge<<<kB * kN, 256, 0, stream>>>(node_feat, Wa, Wb, nfT, e1, e2);
  k_pairwise<<<dim3(kN / 16, kN / 16, kB), 256, 0, stream>>>(
      e1, e2, be, W2e, b2e, adjs, wadj, partials);
  k_rowsum<<<kB * kN, 256, 0, stream>>>(wadj, sums);
  k_scale_f16<<<(kB * kN * kN) / 256, 256, 0, stream>>>(wadj, sums, A16);
  k_wconvT<<<(kD * kH) / 256, 256, 0, stream>>>(Wconv, WconvT);
  k_gemm_agg<<<dim3(kN / 64, kD / 32, kB), 256, 0, stream>>>(A16, nfT, agg16);
  k_gemm_conv<<<dim3(kN / 64, kH / 32, kB), 256, 0, stream>>>(agg16, WconvT, bconv, hbuf);
  k_reduce_pen<<<1, 256, 0, stream>>>(partials, (kN / 16) * (kN / 16) * kB, pen);
  k_pool<<<kB, 256, 0, stream>>>(sums, masks, hbuf, gr);
  k_classifier<<<1, 128, 0, stream>>>(gr, W1c, b1c, ln_g, ln_b, W2c, b2c, u_sn,
                                      labels, pen, out);
}
